// CounterPropagationNetwork_85650237817447
// MI455X (gfx1250) — matrix operand order repaired and checked by reading, not compiled, not run on hardware
//
#include <hip/hip_runtime.h>

// ---------------------------------------------------------------------------
// CounterPropagationNetwork forward for MI455X (gfx1250, wave32, WMMA)
//   x:  [8192, 512]  f32
//   kW: [4096, 512]  f32   (kohonen)
//   gW: [1024, 4096] f32   (grossberg)
// out: [8192,1024] f32  ++  [8192] int32 winners  (concatenated in d_out)
//
// Block = 32 batch rows (two 16-row A tiles) so every B fragment pair feeds
// 6 WMMAs instead of 3 -> halves VMEM loads per WMMA and halves kohonen L2
// re-read traffic. x-tile (bf16 hi/lo) lives in 68.7KB dynamic LDS (CDNA5
// WGP has 320KB).
// ---------------------------------------------------------------------------

#define BATCH   8192
#define INSZ    512
#define HID     4096
#define OUTSZ   1024
#define MROWS   32                    // batch rows per block
#define ROWPAD  (INSZ + 8)            // 1040B row stride -> conflict-free b128

typedef __attribute__((ext_vector_type(16))) __bf16 v16bf;
typedef __attribute__((ext_vector_type(8)))  __bf16 v8bf;
typedef __attribute__((ext_vector_type(8)))  float  v8f;

union BFrag { v16bf v; v8bf h[2]; };

__device__ __forceinline__ unsigned short f32_to_bf16_rne(float f) {
    unsigned u = __float_as_uint(f);
    unsigned r = u + 0x7FFFu + ((u >> 16) & 1u);   // round-to-nearest-even
    return (unsigned short)(r >> 16);
}
__device__ __forceinline__ float bf16_bits_to_f32(unsigned short s) {
    return __uint_as_float(((unsigned)s) << 16);
}

// LDS carve-out (dynamic): sizes in bytes
#define OFF_XHI   0
#define OFF_XLO   (MROWS * ROWPAD * 2)                 // 33280
#define OFF_X2S   (OFF_XLO + MROWS * ROWPAD * 2)       // 66560
#define OFF_REDV  (OFF_X2S + MROWS * 4)                // 66688
#define OFF_REDI  (OFF_REDV + 8 * MROWS * 4)           // 67712
#define LDS_BYTES (OFF_REDI + 8 * MROWS * 4)           // 68736

// ---------------------------------------------------------------------------
// Kernel 1: kohonen f32 -> bf16 (hi) + bf16 (residual lo), plus exact ||w||^2
// ---------------------------------------------------------------------------
__global__ __launch_bounds__(256)
void cpn_prep_kohonen(const float* __restrict__ kw,
                      unsigned short* __restrict__ khi,
                      unsigned short* __restrict__ klo,
                      float* __restrict__ w2) {
    __shared__ float sred[256];
    const int h   = blockIdx.x;
    const int tid = threadIdx.x;
    const float* row = kw + (size_t)h * INSZ;
    float acc = 0.f;
    for (int c = tid; c < INSZ; c += 256) {
        float v  = row[c];
        unsigned short hb = f32_to_bf16_rne(v);
        float hf = bf16_bits_to_f32(hb);
        unsigned short lb = f32_to_bf16_rne(v - hf);
        khi[(size_t)h * INSZ + c] = hb;
        klo[(size_t)h * INSZ + c] = lb;
        acc += v * v;
    }
    sred[tid] = acc;
    __syncthreads();
    for (int s = 128; s > 0; s >>= 1) {
        if (tid < s) sred[tid] += sred[tid + s];
        __syncthreads();
    }
    if (tid == 0) w2[h] = sred[0];
}

// ---------------------------------------------------------------------------
// Kernel 2: fused  dist^2 = ||x||^2 + ||w||^2 - 2 x.w^T  +  argmin over H.
// 256 blocks x 256 threads (8 wave32). Wave w covers hidden [w*512, w*512+512)
// for both 16-row A tiles; bf16 hi/lo split -> 6 WMMAs per K-chunk.
// ---------------------------------------------------------------------------
__global__ __launch_bounds__(256)
void cpn_dist_argmin(const float* __restrict__ x,
                     const unsigned short* __restrict__ khi,
                     const unsigned short* __restrict__ klo,
                     const float* __restrict__ w2,
                     int* __restrict__ winners) {
    extern __shared__ __align__(16) unsigned char smem[];
    unsigned short (*xhi)[ROWPAD] = (unsigned short (*)[ROWPAD])(smem + OFF_XHI);
    unsigned short (*xlo)[ROWPAD] = (unsigned short (*)[ROWPAD])(smem + OFF_XLO);
    float* x2s          = (float*)(smem + OFF_X2S);
    float (*redV)[MROWS] = (float (*)[MROWS])(smem + OFF_REDV);
    int   (*redI)[MROWS] = (int   (*)[MROWS])(smem + OFF_REDI);

    const int tid = threadIdx.x;
    const int b0  = blockIdx.x * MROWS;

    if (tid < MROWS) x2s[tid] = 0.f;
    __syncthreads();

    // ---- load + convert x tile [32 x 512] f32 -> bf16 hi/lo, accumulate ||x||^2
    {
        const int row = tid >> 3;            // 0..31  (8 threads per row)
        const int cb  = (tid & 7) * 64;      // 64 cols each
        const float4* src =
            reinterpret_cast<const float4*>(x + (size_t)(b0 + row) * INSZ + cb);
        float ss = 0.f;
        #pragma unroll
        for (int i = 0; i < 16; ++i) {
            float4 v = src[i];
            float f[4] = {v.x, v.y, v.z, v.w};
            #pragma unroll
            for (int j = 0; j < 4; ++j) {
                float fv = f[j];
                unsigned short hb = f32_to_bf16_rne(fv);
                float hf = bf16_bits_to_f32(hb);
                unsigned short lb = f32_to_bf16_rne(fv - hf);
                int c = cb + i * 4 + j;
                xhi[row][c] = hb;
                xlo[row][c] = lb;
                ss += fv * fv;
            }
        }
        atomicAdd(&x2s[row], ss);            // ds_add_f32
    }
    __syncthreads();

    const int wave = tid >> 5;
    const int lane = tid & 31;
    const int half = lane >> 4;              // 0: lanes 0-15, 1: lanes 16-31
    const int ln   = lane & 15;
    const int kbA  = half * 8;               // A frag: K {0-7,16-23} / {8-15,24-31}
    const int kbB  = half * 8;               // B frag: K {0-7,16-23} / {8-15,24-31}

    float minv0[8], minv1[8];
    int   mini0[8], mini1[8];
    #pragma unroll
    for (int r = 0; r < 8; ++r) {
        minv0[r] = 3.4e38f; mini0[r] = 0x7fffffff;
        minv1[r] = 3.4e38f; mini1[r] = 0x7fffffff;
    }

    for (int t = 0; t < 32; ++t) {
        const int n  = wave * 512 + t * 16 + ln;   // this lane's hidden unit
        const unsigned short* bhp = khi + (size_t)n * INSZ;
        const unsigned short* blp = klo + (size_t)n * INSZ;

        if (t < 31) {                               // warm next tile's rows
            __builtin_prefetch(bhp + 16 * INSZ, 0, 0);
            __builtin_prefetch(blp + 16 * INSZ, 0, 0);
        }

        v8f acc0 = {0.f, 0.f, 0.f, 0.f, 0.f, 0.f, 0.f, 0.f};
        v8f acc1 = {0.f, 0.f, 0.f, 0.f, 0.f, 0.f, 0.f, 0.f};
        #pragma unroll 2
        for (int kc = 0; kc < 16; ++kc) {
            const int koA = kc * 32 + kbA;
            const int koB = kc * 32 + kbB;
            BFrag ah0, al0, ah1, al1, bh, bl;
            bh.h[0] = *reinterpret_cast<const v8bf*>(bhp + koB);
            bh.h[1] = *reinterpret_cast<const v8bf*>(bhp + koB + 16);
            bl.h[0] = *reinterpret_cast<const v8bf*>(blp + koB);
            bl.h[1] = *reinterpret_cast<const v8bf*>(blp + koB + 16);
            ah0.h[0] = *reinterpret_cast<const v8bf*>(&xhi[ln][koA]);
            ah0.h[1] = *reinterpret_cast<const v8bf*>(&xhi[ln][koA + 16]);
            al0.h[0] = *reinterpret_cast<const v8bf*>(&xlo[ln][koA]);
            al0.h[1] = *reinterpret_cast<const v8bf*>(&xlo[ln][koA + 16]);
            ah1.h[0] = *reinterpret_cast<const v8bf*>(&xhi[16 + ln][koA]);
            ah1.h[1] = *reinterpret_cast<const v8bf*>(&xhi[16 + ln][koA + 16]);
            al1.h[0] = *reinterpret_cast<const v8bf*>(&xlo[16 + ln][koA]);
            al1.h[1] = *reinterpret_cast<const v8bf*>(&xlo[16 + ln][koA + 16]);
            // rows 0-15
            acc0 = __builtin_amdgcn_wmma_f32_16x16x32_bf16(
                       false, ah0.v, false, bh.v, (short)0, acc0, false, false);
            acc0 = __builtin_amdgcn_wmma_f32_16x16x32_bf16(
                       false, ah0.v, false, bl.v, (short)0, acc0, false, false);
            acc0 = __builtin_amdgcn_wmma_f32_16x16x32_bf16(
                       false, al0.v, false, bh.v, (short)0, acc0, false, false);
            // rows 16-31 (reuse B fragments)
            acc1 = __builtin_amdgcn_wmma_f32_16x16x32_bf16(
                       false, ah1.v, false, bh.v, (short)0, acc1, false, false);
            acc1 = __builtin_amdgcn_wmma_f32_16x16x32_bf16(
                       false, ah1.v, false, bl.v, (short)0, acc1, false, false);
            acc1 = __builtin_amdgcn_wmma_f32_16x16x32_bf16(
                       false, al1.v, false, bh.v, (short)0, acc1, false, false);
        }

        const float wn = w2[n];
        #pragma unroll
        for (int r = 0; r < 8; ++r) {
            const int m = r + half * 8;          // C layout: lanes 16-31 -> M+8
            float d2a = fmaxf(x2s[m] + wn - 2.0f * acc0[r], 0.0f);
            if (d2a < minv0[r]) { minv0[r] = d2a; mini0[r] = n; }  // n increasing
            float d2b = fmaxf(x2s[16 + m] + wn - 2.0f * acc1[r], 0.0f);
            if (d2b < minv1[r]) { minv1[r] = d2b; mini1[r] = n; }
        }
    }

    // ---- reduce within each 16-lane half (xor masks < 16 stay in the half)
    #pragma unroll
    for (int r = 0; r < 8; ++r) {
        float v0 = minv0[r], v1 = minv1[r];
        int   i0 = mini0[r], i1 = mini1[r];
        #pragma unroll
        for (int off = 1; off < 16; off <<= 1) {
            float ov0 = __shfl_xor(v0, off, 32);
            int   oi0 = __shfl_xor(i0, off, 32);
            if (ov0 < v0 || (ov0 == v0 && oi0 < i0)) { v0 = ov0; i0 = oi0; }
            float ov1 = __shfl_xor(v1, off, 32);
            int   oi1 = __shfl_xor(i1, off, 32);
            if (ov1 < v1 || (ov1 == v1 && oi1 < i1)) { v1 = ov1; i1 = oi1; }
        }
        if (ln == 0) {
            const int m = r + half * 8;
            redV[wave][m]      = v0;
            redI[wave][m]      = i0;
            redV[wave][16 + m] = v1;
            redI[wave][16 + m] = i1;
        }
    }
    __syncthreads();

    // ---- cross-wave reduce, lowest-index tie-break (matches jnp.argmin)
    if (tid < MROWS) {
        float v = redV[0][tid];
        int   i = redI[0][tid];
        #pragma unroll
        for (int w = 1; w < 8; ++w) {
            float ov = redV[w][tid];
            int   oi = redI[w][tid];
            if (ov < v || (ov == v && oi < i)) { v = ov; i = oi; }
        }
        winners[b0 + tid] = i;
    }
}

// ---------------------------------------------------------------------------
// Kernel 3: output[b][o] = grossberg[o][winner[b]]  (gW is 16MB -> L2-resident)
// ---------------------------------------------------------------------------
__global__ __launch_bounds__(256)
void cpn_gather(const float* __restrict__ gw,
                const int* __restrict__ winners,
                float* __restrict__ out) {
    const int b   = blockIdx.x;
    const int win = winners[b];
    #pragma unroll
    for (int o = threadIdx.x; o < OUTSZ; o += 256)
        out[(size_t)b * OUTSZ + o] = gw[(size_t)o * HID + win];
}

// ---------------------------------------------------------------------------
extern "C" void kernel_launch(void* const* d_in, const int* in_sizes, int n_in,
                              void* d_out, int out_size, void* d_ws, size_t ws_size,
                              hipStream_t stream) {
    const float* x  = (const float*)d_in[0];
    const float* kw = (const float*)d_in[1];
    const float* gw = (const float*)d_in[2];

    float* out     = (float*)d_out;
    int*   winners = (int*)(out + (size_t)BATCH * OUTSZ);

    unsigned char*  ws  = (unsigned char*)d_ws;
    unsigned short* khi = (unsigned short*)ws;                              // 4 MiB
    unsigned short* klo = (unsigned short*)(ws + (size_t)HID * INSZ * 2);   // 4 MiB
    float*          w2  = (float*)(ws + (size_t)HID * INSZ * 4);            // 16 KiB

    cpn_prep_kohonen<<<HID,           256, 0,         stream>>>(kw, khi, klo, w2);
    cpn_dist_argmin <<<BATCH / MROWS, 256, LDS_BYTES, stream>>>(x, khi, klo, w2, winners);
    cpn_gather      <<<BATCH,         256, 0,         stream>>>(gw, winners, out);
}
